// Model_39676907885555
// MI455X (gfx1250) — matrix operand order repaired and checked by reading, not compiled, not run on hardware
//
#include <hip/hip_runtime.h>
#include <hip/hip_bf16.h>
#include <cstdint>
#include <cstddef>

// ---------------------------------------------------------------- types
typedef __bf16 v16bf __attribute__((ext_vector_type(16)));
typedef float  v8f   __attribute__((ext_vector_type(8)));

union FragBF { v16bf v; uint4 u[2]; };

// Problem dims
#define BB 8
#define SS 2048
#define DD 1024
#define H1D 512
#define DHD 128
#define DVD 512
#define DOUTD 512
#define MROWS (BB * SS)   // 16384

// ---------------------------------------------------------------- WMMA helpers
// A fragment: 16x32 (MxK), row-major source with leading dim lda (elements).
// Per ISA 7.12.2: lanes 0-15 -> M=lane, K in {kb+0..7, kb+16..23}, kb=0;
//                 lanes 16-31 -> M=lane-16, same pattern with kb=8.
__device__ __forceinline__ v16bf load_a_rm(const __bf16* p0, int lda) {
  const int l = threadIdx.x & 31;
  const __bf16* p = p0 + (size_t)(l & 15) * lda + ((l >> 4) << 3);
  FragBF f;
  f.u[0] = *(const uint4*)(p);       // K = kb+0 .. kb+7
  f.u[1] = *(const uint4*)(p + 16);  // K = kb+16 .. kb+23
  return f.v;
}

// B fragment: 32x16 (KxN), source stored "column major" i.e. element (k,n) at
// p0[n*ldn + k]. Lanes 0-15: N=lane, K={0..7,16..23} ; lanes 16-31: N=lane-16, K={8..15,24..31}.
__device__ __forceinline__ v16bf load_b_cm(const __bf16* p0, int ldn) {
  const int l = threadIdx.x & 31;
  const __bf16* p = p0 + (size_t)(l & 15) * ldn + ((l >> 4) << 3);
  FragBF f;
  f.u[0] = *(const uint4*)(p);
  f.u[1] = *(const uint4*)(p + 16);
  return f.v;
}

__device__ __forceinline__ v8f wmma_bf16(v16bf a, v16bf b, v8f c) {
  return __builtin_amdgcn_wmma_f32_16x16x32_bf16(
      /*neg_a=*/false, a, /*neg_b=*/false, b,
      /*c_mod=*/(short)0, c, /*reuse_a=*/false, /*reuse_b=*/false);
}

// C/D 16x16 f32: lane n = l&15, VGPR r holds row M = r + 8*(l>=16)
__device__ __forceinline__ void store_c(float* C, int ldc, v8f c) {
  const int l = threadIdx.x & 31;
  float* p = C + (size_t)((l >> 4) * 8) * ldc + (l & 15);
  #pragma unroll
  for (int r = 0; r < 8; ++r) p[(size_t)r * ldc] = c[r];
}

// ---------------------------------------------------------------- GEMM
// C[M,N] f32 = A[M,K] bf16 (row-major) @ Bt[N,K] bf16 (weight stored transposed)
// 256 threads = 8 waves arranged 4x2; block tile 128x128; per-wave 32x64
// (2 A-frags reused against 4 B-frags -> 8 WMMAs per k-step).
__global__ __launch_bounds__(256) void gemm_bf16_f32(
    const __bf16* __restrict__ A, const __bf16* __restrict__ Bt,
    float* __restrict__ C, int M, int N, int K) {
  const int wave = threadIdx.x >> 5;
  const int m0 = blockIdx.y * 128 + (wave >> 1) * 32;
  const int n0 = blockIdx.x * 128 + (wave & 1) * 64;
  const __bf16* Ar0 = A + (size_t)m0 * K;
  const __bf16* Ar1 = A + (size_t)(m0 + 16) * K;
  const __bf16* Bc0 = Bt + (size_t)n0 * K;
  const __bf16* Bc1 = Bt + (size_t)(n0 + 16) * K;
  const __bf16* Bc2 = Bt + (size_t)(n0 + 32) * K;
  const __bf16* Bc3 = Bt + (size_t)(n0 + 48) * K;

  v8f acc[2][4];
  #pragma unroll
  for (int i = 0; i < 2; ++i)
    #pragma unroll
    for (int j = 0; j < 4; ++j) acc[i][j] = (v8f){};

  for (int k = 0; k < K; k += 32) {
    if (k + 128 < K) {  // WGP-scope prefetch of upcoming k-tiles (locality 3)
      __builtin_prefetch(Ar0 + k + 128, 0, 3);
      __builtin_prefetch(Ar1 + k + 128, 0, 3);
      __builtin_prefetch(Bc0 + k + 128, 0, 3);
      __builtin_prefetch(Bc2 + k + 128, 0, 3);
    }
    v16bf fa0 = load_a_rm(Ar0 + k, K);
    v16bf fa1 = load_a_rm(Ar1 + k, K);
    v16bf fb0 = load_b_cm(Bc0 + k, K);
    v16bf fb1 = load_b_cm(Bc1 + k, K);
    v16bf fb2 = load_b_cm(Bc2 + k, K);
    v16bf fb3 = load_b_cm(Bc3 + k, K);
    acc[0][0] = wmma_bf16(fa0, fb0, acc[0][0]);
    acc[0][1] = wmma_bf16(fa0, fb1, acc[0][1]);
    acc[0][2] = wmma_bf16(fa0, fb2, acc[0][2]);
    acc[0][3] = wmma_bf16(fa0, fb3, acc[0][3]);
    acc[1][0] = wmma_bf16(fa1, fb0, acc[1][0]);
    acc[1][1] = wmma_bf16(fa1, fb1, acc[1][1]);
    acc[1][2] = wmma_bf16(fa1, fb2, acc[1][2]);
    acc[1][3] = wmma_bf16(fa1, fb3, acc[1][3]);
  }
  #pragma unroll
  for (int i = 0; i < 2; ++i)
    #pragma unroll
    for (int j = 0; j < 4; ++j)
      store_c(C + (size_t)(m0 + 16 * i) * N + n0 + 16 * j, N, acc[i][j]);
}

// ---------------------------------------------------------------- flash attention
// grid: (S/64, B, DV/128); block = 128 threads = 4 waves.
// Wave w handles query rows [64*bx + 16*w, +16), full DH=128, DV slice of 128.
// q2,k3: (B*S,128) bf16 row-major. vt: (B,DV,S) bf16. out: (B*S,512) f32.
__global__ __launch_bounds__(128) void flash_causal(
    const __bf16* __restrict__ q2, const __bf16* __restrict__ k3,
    const __bf16* __restrict__ vt, float* __restrict__ out) {
  __shared__ __align__(16) __bf16 lds_p[4][16 * 32];
  const int lane = threadIdx.x & 31;
  const int wave = threadIdx.x >> 5;
  const int hi = lane >> 4;
  const int nn = lane & 15;
  const int b = blockIdx.y;
  const int dv0 = blockIdx.z * 128;
  const int qbase = blockIdx.x * 64 + wave * 16;

  const __bf16* q2b = q2 + ((size_t)b * SS + qbase) * DHD;
  const __bf16* k3b = k3 + (size_t)b * SS * DHD;
  const __bf16* vtb = vt + ((size_t)b * DVD + dv0) * SS;
  float* outb = out + ((size_t)b * SS + qbase) * DVD + dv0;

  // Q fragments held in registers for the whole pass (16 x 128 -> 4 frags)
  v16bf aq[4];
  #pragma unroll
  for (int t = 0; t < 4; ++t) aq[t] = load_a_rm(q2b + 32 * t, DHD);

  v8f acc[8];
  #pragma unroll
  for (int t = 0; t < 8; ++t) acc[t] = (v8f){};
  float mrow[8], lrow[8];
  #pragma unroll
  for (int r = 0; r < 8; ++r) { mrow[r] = -1.0e30f; lrow[r] = 0.0f; }

  const float scale = 0.088388347648318447f;  // 1/sqrt(128)
  const int nkb = (qbase + 47) >> 5;          // key blocks of 32 up to diagonal

  for (int kb = 0; kb < nkb; ++kb) {
    const int key0 = kb * 32;
    // ---- S = q2 @ k3^T  (16 x 32 slab, two 16x16 tiles)
    v8f s0 = {}, s1 = {};
    #pragma unroll
    for (int t = 0; t < 4; ++t) {
      s0 = wmma_bf16(aq[t], load_b_cm(k3b + (size_t)key0 * DHD + 32 * t, DHD), s0);
      s1 = wmma_bf16(aq[t], load_b_cm(k3b + (size_t)(key0 + 16) * DHD + 32 * t, DHD), s1);
    }
    // ---- scale + causal mask + online softmax
    float alpha[8];
    #pragma unroll
    for (int r = 0; r < 8; ++r) {
      const int qrow = qbase + r + 8 * hi;
      float v0 = s0[r] * scale;
      float v1 = s1[r] * scale;
      if (key0 + nn > qrow)      v0 = -1.0e30f;
      if (key0 + 16 + nn > qrow) v1 = -1.0e30f;
      float mt = fmaxf(v0, v1);
      mt = fmaxf(mt, __shfl_xor(mt, 8, 32));
      mt = fmaxf(mt, __shfl_xor(mt, 4, 32));
      mt = fmaxf(mt, __shfl_xor(mt, 2, 32));
      mt = fmaxf(mt, __shfl_xor(mt, 1, 32));
      const float mnew = fmaxf(mrow[r], mt);
      alpha[r] = __expf(mrow[r] - mnew);
      mrow[r] = mnew;
      const float p0 = __expf(v0 - mnew);
      const float p1 = __expf(v1 - mnew);
      float rs = p0 + p1;
      rs += __shfl_xor(rs, 8, 32);
      rs += __shfl_xor(rs, 4, 32);
      rs += __shfl_xor(rs, 2, 32);
      rs += __shfl_xor(rs, 1, 32);
      lrow[r] = lrow[r] * alpha[r] + rs;
      // stage P (bf16) in LDS, row-major 16x32
      lds_p[wave][(r + 8 * hi) * 32 + nn]      = (__bf16)p0;
      lds_p[wave][(r + 8 * hi) * 32 + 16 + nn] = (__bf16)p1;
    }
    // rescale running output
    #pragma unroll
    for (int t = 0; t < 8; ++t)
      #pragma unroll
      for (int r = 0; r < 8; ++r) acc[t][r] *= alpha[r];

    asm volatile("s_wait_dscnt 0" ::: "memory");  // P visible before re-read

    // ---- O += P @ V  (A = P from LDS; B = V^T slice, contiguous along keys)
    v16bf ap = load_a_rm(&lds_p[wave][0], 32);
    #pragma unroll
    for (int t = 0; t < 8; ++t)
      acc[t] = wmma_bf16(ap, load_b_cm(vtb + (size_t)(16 * t) * SS + key0, SS), acc[t]);
  }

  // ---- normalize + store
  float inv[8];
  #pragma unroll
  for (int r = 0; r < 8; ++r) inv[r] = 1.0f / lrow[r];
  #pragma unroll
  for (int t = 0; t < 8; ++t)
    #pragma unroll
    for (int r = 0; r < 8; ++r)
      outb[(size_t)(r + 8 * hi) * DVD + 16 * t + nn] = acc[t][r] * inv[r];
}

// ---------------------------------------------------------------- elementwise
__global__ void cvt_bf16_k(const float* __restrict__ in, __bf16* __restrict__ out,
                           long long n) {
  long long i = (long long)blockIdx.x * blockDim.x + threadIdx.x;
  const long long st = (long long)gridDim.x * blockDim.x;
  for (; i < n; i += st) out[i] = (__bf16)in[i];
}

__global__ void addcvt_bf16_k(const float* __restrict__ a, const float* __restrict__ b,
                              __bf16* __restrict__ out, long long n) {
  long long i = (long long)blockIdx.x * blockDim.x + threadIdx.x;
  const long long st = (long long)gridDim.x * blockDim.x;
  for (; i < n; i += st) out[i] = (__bf16)(a[i] + b[i]);
}

// weight (K,N) f32 -> (N,K) bf16
__global__ void wtrans_k(const float* __restrict__ in, __bf16* __restrict__ out,
                         int K, int N) {
  long long i = (long long)blockIdx.x * blockDim.x + threadIdx.x;
  const long long n = (long long)K * N;
  const long long st = (long long)gridDim.x * blockDim.x;
  for (; i < n; i += st) {
    const int r = (int)(i / N), c = (int)(i % N);
    out[(size_t)c * K + r] = (__bf16)in[i];
  }
}

// w7sum^T: out[n*512+k] = bf16(w7[k][n] + w7[k+512][n]); w7 is (1024,512)
__global__ void w7prep_k(const float* __restrict__ w7, __bf16* __restrict__ out) {
  long long i = (long long)blockIdx.x * blockDim.x + threadIdx.x;
  const long long n = (long long)DVD * DOUTD;
  const long long st = (long long)gridDim.x * blockDim.x;
  for (; i < n; i += st) {
    const int k = (int)(i / DOUTD), c = (int)(i % DOUTD);
    out[(size_t)c * DVD + k] =
        (__bf16)(w7[(size_t)k * DOUTD + c] + w7[(size_t)(k + DVD) * DOUTD + c]);
  }
}

// v (B,S,DV) f32 -> vt (B,DV,S) bf16
__global__ void vtrans_k(const float* __restrict__ v, __bf16* __restrict__ vt) {
  long long i = (long long)blockIdx.x * blockDim.x + threadIdx.x;
  const long long n = (long long)BB * SS * DVD;
  const long long st = (long long)gridDim.x * blockDim.x;
  for (; i < n; i += st) {
    const int b = (int)(i / ((long long)SS * DVD));
    const int rem = (int)(i % ((long long)SS * DVD));
    const int s = rem / DVD, dv = rem % DVD;
    vt[((size_t)b * DVD + dv) * SS + s] = (__bf16)v[i];
  }
}

// ---------------------------------------------------------------- launcher
extern "C" void kernel_launch(void* const* d_in, const int* in_sizes, int n_in,
                              void* d_out, int out_size, void* d_ws, size_t ws_size,
                              hipStream_t stream) {
  (void)in_sizes; (void)n_in; (void)out_size; (void)ws_size;
  const float* q  = (const float*)d_in[0];
  const float* k  = (const float*)d_in[1];
  const float* v  = (const float*)d_in[2];
  const float* w1 = (const float*)d_in[3];
  const float* w2 = (const float*)d_in[4];
  const float* w3 = (const float*)d_in[5];
  const float* w4 = (const float*)d_in[6];
  const float* w6 = (const float*)d_in[7];
  const float* w7 = (const float*)d_in[8];

  size_t off = 0;
  auto alloc = [&](size_t bytes) {
    void* p = (char*)d_ws + off;
    off += (bytes + 255) & ~(size_t)255;
    return p;
  };
  __bf16* qbf   = (__bf16*)alloc((size_t)MROWS * DD * 2);
  __bf16* kbf   = (__bf16*)alloc((size_t)MROWS * DD * 2);
  __bf16* vtbf  = (__bf16*)alloc((size_t)BB * DVD * SS * 2);
  __bf16* w1t   = (__bf16*)alloc((size_t)DD * H1D * 2);
  __bf16* w2t   = (__bf16*)alloc((size_t)H1D * DHD * 2);
  __bf16* w3t   = (__bf16*)alloc((size_t)DD * DVD * 2);
  __bf16* w4t   = (__bf16*)alloc((size_t)DD * H1D * 2);
  __bf16* w6t   = (__bf16*)alloc((size_t)H1D * DHD * 2);
  __bf16* w7t   = (__bf16*)alloc((size_t)DVD * DOUTD * 2);
  float*  tf32  = (float*)alloc((size_t)MROWS * H1D * 4);
  __bf16* tbf   = (__bf16*)alloc((size_t)MROWS * H1D * 2);
  float*  smallf= (float*)alloc((size_t)MROWS * DHD * 4);
  __bf16* q2bf  = (__bf16*)alloc((size_t)MROWS * DHD * 2);
  __bf16* k3bf  = (__bf16*)alloc((size_t)MROWS * DHD * 2);
  float*  q3f   = (float*)alloc((size_t)MROWS * DVD * 4);
  float*  outf  = (float*)alloc((size_t)MROWS * DVD * 4);
  __bf16* sumbf = (__bf16*)alloc((size_t)MROWS * DVD * 2);

  const int EB = 256, EG = 4096;
  // input / weight preparation
  cvt_bf16_k<<<EG, EB, 0, stream>>>(q, qbf, (long long)MROWS * DD);
  cvt_bf16_k<<<EG, EB, 0, stream>>>(k, kbf, (long long)MROWS * DD);
  vtrans_k  <<<EG, EB, 0, stream>>>(v, vtbf);
  wtrans_k  <<<EG, EB, 0, stream>>>(w1, w1t, DD, H1D);
  wtrans_k  <<<EG, EB, 0, stream>>>(w2, w2t, H1D, DHD);
  wtrans_k  <<<EG, EB, 0, stream>>>(w3, w3t, DD, DVD);
  wtrans_k  <<<EG, EB, 0, stream>>>(w4, w4t, DD, H1D);
  wtrans_k  <<<EG, EB, 0, stream>>>(w6, w6t, H1D, DHD);
  w7prep_k  <<<EG, EB, 0, stream>>>(w7, w7t);

  // q2 = (q @ w1) @ w2
  gemm_bf16_f32<<<dim3(H1D / 128, MROWS / 128), 256, 0, stream>>>(qbf, w1t, tf32, MROWS, H1D, DD);
  cvt_bf16_k<<<EG, EB, 0, stream>>>(tf32, tbf, (long long)MROWS * H1D);
  gemm_bf16_f32<<<dim3(1, MROWS / 128), 256, 0, stream>>>(tbf, w2t, smallf, MROWS, DHD, H1D);
  cvt_bf16_k<<<EG, EB, 0, stream>>>(smallf, q2bf, (long long)MROWS * DHD);
  // q3 = q @ w3   (kept f32 for the final add)
  gemm_bf16_f32<<<dim3(DVD / 128, MROWS / 128), 256, 0, stream>>>(qbf, w3t, q3f, MROWS, DVD, DD);
  // k3 = (k @ w4) @ w6
  gemm_bf16_f32<<<dim3(H1D / 128, MROWS / 128), 256, 0, stream>>>(kbf, w4t, tf32, MROWS, H1D, DD);
  cvt_bf16_k<<<EG, EB, 0, stream>>>(tf32, tbf, (long long)MROWS * H1D);
  gemm_bf16_f32<<<dim3(1, MROWS / 128), 256, 0, stream>>>(tbf, w6t, smallf, MROWS, DHD, H1D);
  cvt_bf16_k<<<EG, EB, 0, stream>>>(smallf, k3bf, (long long)MROWS * DHD);

  // causal attention: out = softmax(q2 k3^T / sqrt(DH)) @ v
  flash_causal<<<dim3(SS / 64, BB, DVD / 128), 128, 0, stream>>>(q2bf, k3bf, vtbf, outf);

  // final: d_out = (out + q3) @ (w7a + w7b)
  addcvt_bf16_k<<<EG, EB, 0, stream>>>(outf, q3f, sumbf, (long long)MROWS * DVD);
  gemm_bf16_f32<<<dim3(DOUTD / 128, MROWS / 128), 256, 0, stream>>>(sumbf, w7t, (float*)d_out,
                                                                    MROWS, DOUTD, DVD);
}
